// HypercomplexLinear_73375221284962
// MI455X (gfx1250) — matrix operand order repaired and checked by reading, not compiled, not run on hardware
//
#include <hip/hip_runtime.h>

// ---------------------------------------------------------------------------
// Hypercomplex linear (N=4 block-circulant) for gfx1250:
//   C[6272,3072] = X[6272,768] * Weff[768,3072] + bias
//   Weff[j*192+c, kblk*768+o] = W[(kblk-j)&3, o, c]
// Strategy:
//   prep1: X  -> bf16 hi/lo planes           [6272][768]  (one-time split)
//   prep2: W  -> circulant-permuted Weff as bf16 hi/lo planes, stored
//          N-major/K-contiguous [3072][768]  (moves all index math + cvt
//          out of the GEMM hot loop)
//   gemm : double-buffered LDS tiles filled with GLOBAL_LOAD_ASYNC_TO_LDS_B128
//          (ASYNCcnt-tracked) ; 3-product split-bf16 WMMA
//          (ah*bh + ah*bl + al*bh) for ~fp32 accuracy at 3/8 the
//          instruction cost of exact V_WMMA_F32_16X16X4_F32.
// ---------------------------------------------------------------------------

typedef __attribute__((ext_vector_type(16))) __bf16 v16bf;
typedef __attribute__((ext_vector_type(8)))  __bf16 bf16x8;
typedef __attribute__((ext_vector_type(4)))  __bf16 bf16x4;
typedef __attribute__((ext_vector_type(8)))  float  v8f;
typedef __attribute__((ext_vector_type(4)))  float  f32x4;
typedef int i32x4 __attribute__((vector_size(16)));   // matches builtin param type

#define M_TOTAL 6272   // 32*196
#define N_TOTAL 3072
#define K_TOTAL 768
#define BM 128
#define BN 128
#define BK 32
#define LDK 40         // padded LDS K-stride (bf16): 80B rows, breaks bank conflicts
#define KSTEPS (K_TOTAL / BK)

#if __has_builtin(__builtin_amdgcn_global_load_async_to_lds_b128)
#define USE_ASYNC 1
#else
#define USE_ASYNC 0
#endif

// ---- 16B tile-chunk copy: global bf16 plane -> LDS --------------------------
static __device__ __forceinline__ void copy16(const __bf16* g, __bf16* l) {
#if USE_ASYNC
    __builtin_amdgcn_global_load_async_to_lds_b128(
        (__attribute__((address_space(1))) i32x4*)g,
        (__attribute__((address_space(3))) i32x4*)l,
        0, 0);
#else
    *(bf16x8*)l = *(const bf16x8*)g;
#endif
}

static __device__ __forceinline__ void wait_async_le8() {
#if USE_ASYNC
#if __has_builtin(__builtin_amdgcn_s_wait_asynccnt)
    __builtin_amdgcn_s_wait_asynccnt(8);
#else
    asm volatile("s_wait_asynccnt 0x8" ::: "memory");
#endif
#endif
}

static __device__ __forceinline__ void wait_async_le0() {
#if USE_ASYNC
#if __has_builtin(__builtin_amdgcn_s_wait_asynccnt)
    __builtin_amdgcn_s_wait_asynccnt(0);
#else
    asm volatile("s_wait_asynccnt 0x0" ::: "memory");
#endif
#endif
}

// Load a wave32 WMMA bf16 operand fragment: two contiguous 8-element (16B)
// chunks at 16B-aligned offsets o0 / o1 within one K-contiguous LDS row.
static __device__ __forceinline__ v16bf ld_frag(const __bf16* p, int o0, int o1) {
    bf16x8 a = *(const bf16x8*)(p + o0);
    bf16x8 b = *(const bf16x8*)(p + o1);
    return __builtin_shufflevector(a, b, 0, 1, 2, 3, 4, 5, 6, 7,
                                          8, 9, 10, 11, 12, 13, 14, 15);
}

// ---- prep 1: split X (fp32) into bf16 hi/lo planes --------------------------
__global__ __launch_bounds__(256)
void split_x_kernel(const float* __restrict__ X,
                    __bf16* __restrict__ Xhi, __bf16* __restrict__ Xlo) {
    const size_t idx = ((size_t)blockIdx.x * 256 + threadIdx.x) * 4;
    const f32x4 v = *(const f32x4*)(X + idx);
    bf16x4 h, l;
    #pragma unroll
    for (int e = 0; e < 4; ++e) {
        const __bf16 hb = (__bf16)v[e];          // RNE
        h[e] = hb;
        l[e] = (__bf16)(v[e] - (float)hb);       // residual
    }
    *(bf16x4*)(Xhi + idx) = h;
    *(bf16x4*)(Xlo + idx) = l;
}

// ---- prep 2: build circulant-permuted Weff[3072][768] as bf16 hi/lo ---------
__global__ __launch_bounds__(256)
void build_w_kernel(const float* __restrict__ W,
                    __bf16* __restrict__ Whi, __bf16* __restrict__ Wlo) {
    const size_t idx = ((size_t)blockIdx.x * 256 + threadIdx.x) * 4; // elem in [3072][768]
    const int n = (int)(idx / K_TOTAL);       // output column 0..3071
    const int k = (int)(idx - (size_t)n * K_TOTAL);
    const int kblk = n / 768;                 // hypercomplex output index
    const int o    = n - kblk * 768;
    const int j    = k / 192;                 // hypercomplex input index (4 | k-chunk)
    const int c    = k - j * 192;
    const int ic   = (kblk - j) & 3;          // circulant component select
    const f32x4 v = *(const f32x4*)(W + (size_t)(ic * 768 + o) * 192 + c);
    bf16x4 h, l;
    #pragma unroll
    for (int e = 0; e < 4; ++e) {
        const __bf16 hb = (__bf16)v[e];
        h[e] = hb;
        l[e] = (__bf16)(v[e] - (float)hb);
    }
    *(bf16x4*)(Whi + idx) = h;
    *(bf16x4*)(Wlo + idx) = l;
}

// Issue one 128x32 bf16 tile copy (8KB): 512 x 16B chunks, 2 per thread.
static __device__ __forceinline__
void issue_tile(const __bf16* __restrict__ gbase,  // plane + row0*768 + k0
                __bf16 (*lds)[LDK], int tid) {
    #pragma unroll
    for (int i = 0; i < 2; ++i) {
        const int chunk = tid + i * 256;   // 0..511
        const int row   = chunk >> 2;      // 0..127
        const int q     = chunk & 3;       // 16B chunk within 64B row
        copy16(gbase + (size_t)row * K_TOTAL + q * 8, &lds[row][q * 8]);
    }
}

// ---- main GEMM --------------------------------------------------------------
__global__ __launch_bounds__(256)
void hc_gemm_kernel(const __bf16* __restrict__ Xhi, const __bf16* __restrict__ Xlo,
                    const __bf16* __restrict__ Whi, const __bf16* __restrict__ Wlo,
                    const float* __restrict__ bias, float* __restrict__ Out)
{
    // double-buffered [buf][hi/lo] tiles: 2*2*128*40*2B * 2 arrays = 80 KB
    __shared__ __attribute__((aligned(16))) __bf16 sX[2][2][BM][LDK];
    __shared__ __attribute__((aligned(16))) __bf16 sW[2][2][BN][LDK];

    const int tid  = threadIdx.x;
    const int lane = tid & 31;
    const int wave = tid >> 5;      // 8 waves
    const int wm   = wave >> 1;     // 0..3  -> 32-row M sub-block
    const int wn   = wave & 1;      // 0..1  -> 64-col N sub-block
    const int lrow = lane & 15;
    const int lhi  = (lane >> 4) & 1;

    const int m0 = blockIdx.y * BM;
    const int n0 = blockIdx.x * BN;

    const __bf16* xh0 = Xhi + (size_t)m0 * K_TOTAL;
    const __bf16* xl0 = Xlo + (size_t)m0 * K_TOTAL;
    const __bf16* wh0 = Whi + (size_t)n0 * K_TOTAL;
    const __bf16* wl0 = Wlo + (size_t)n0 * K_TOTAL;

    v8f acc[2][4];
    #pragma unroll
    for (int mi = 0; mi < 2; ++mi)
        #pragma unroll
        for (int ni = 0; ni < 4; ++ni)
            acc[mi][ni] = (v8f)0.0f;

    // prologue: start tile 0
    issue_tile(xh0, sX[0][0], tid);
    issue_tile(xl0, sX[0][1], tid);
    issue_tile(wh0, sW[0][0], tid);
    issue_tile(wl0, sW[0][1], tid);

    for (int ks = 0; ks < KSTEPS; ++ks) {
        const int buf = ks & 1;
        if (ks + 1 < KSTEPS) {
            const int k1 = (ks + 1) * BK;
            issue_tile(xh0 + k1, sX[buf ^ 1][0], tid);
            issue_tile(xl0 + k1, sX[buf ^ 1][1], tid);
            issue_tile(wh0 + k1, sW[buf ^ 1][0], tid);
            issue_tile(wl0 + k1, sW[buf ^ 1][1], tid);
            wait_async_le8();   // drain current tile's 8 async ops (in-order)
        } else {
            wait_async_le0();
        }
        __syncthreads();        // all waves' copies of buf are visible

        // A frag (16x32, ISA 7.12.2): lanes 0-15 K{0..7,16..23}, lanes 16-31
        // K{8..15,24..31} of row M=lane&15.
        v16bf ahi[2], alo[2];
        const int akoff = lhi * 8;
        #pragma unroll
        for (int mi = 0; mi < 2; ++mi) {
            const int r = wm * 32 + mi * 16 + lrow;
            ahi[mi] = ld_frag(&sX[buf][0][r][0], akoff, akoff + 16);
            alo[mi] = ld_frag(&sX[buf][1][r][0], akoff, akoff + 16);
        }
        // B frag (32x16): lanes 0-15 hold K{0..7,16..23}, lanes 16-31
        // K{8..15,24..31} of column N=lane&15.
        const int bkoff = lhi * 8;
        #pragma unroll
        for (int ni = 0; ni < 4; ++ni) {
            const int r = wn * 64 + ni * 16 + lrow;
            const v16bf bhi = ld_frag(&sW[buf][0][r][0], bkoff, bkoff + 16);
            const v16bf blo = ld_frag(&sW[buf][1][r][0], bkoff, bkoff + 16);
            #pragma unroll
            for (int mi = 0; mi < 2; ++mi) {
                acc[mi][ni] = __builtin_amdgcn_wmma_f32_16x16x32_bf16(
                    false, ahi[mi], false, bhi, (short)0, acc[mi][ni], false, false);
                acc[mi][ni] = __builtin_amdgcn_wmma_f32_16x16x32_bf16(
                    false, alo[mi], false, bhi, (short)0, acc[mi][ni], false, false);
                acc[mi][ni] = __builtin_amdgcn_wmma_f32_16x16x32_bf16(
                    false, ahi[mi], false, blo, (short)0, acc[mi][ni], false, false);
            }
        }
        __syncthreads();        // done reading buf before it is overwritten
    }

    // epilogue: C/D layout (VGPR v: M = v + 8*lhi, N = lane&15)
    #pragma unroll
    for (int ni = 0; ni < 4; ++ni) {
        const int col = n0 + wn * 64 + ni * 16 + lrow;
        const float bv = bias[col];
        #pragma unroll
        for (int mi = 0; mi < 2; ++mi) {
            const int rbase = m0 + wm * 32 + mi * 16 + lhi * 8;
            #pragma unroll
            for (int v = 0; v < 8; ++v) {
                Out[(size_t)(rbase + v) * N_TOTAL + col] = acc[mi][ni][v] + bv;
            }
        }
    }
}

extern "C" void kernel_launch(void* const* d_in, const int* in_sizes, int n_in,
                              void* d_out, int out_size, void* d_ws, size_t ws_size,
                              hipStream_t stream) {
    (void)in_sizes; (void)n_in; (void)out_size; (void)ws_size;
    const float* X    = (const float*)d_in[0];  // [32,196,768]
    const float* W    = (const float*)d_in[1];  // [4,768,192]
    const float* bias = (const float*)d_in[2];  // [3072]
    float* Out        = (float*)d_out;          // [32,196,3072]

    // workspace layout (bf16): Xhi, Xlo [6272*768] ; Whi, Wlo [3072*768]
    const size_t xplane = (size_t)M_TOTAL * K_TOTAL;   // 4,816,896
    const size_t wplane = (size_t)N_TOTAL * K_TOTAL;   // 2,359,296
    __bf16* Xhi = (__bf16*)d_ws;                       // total ~27.4 MB used
    __bf16* Xlo = Xhi + xplane;
    __bf16* Whi = Xlo + xplane;
    __bf16* Wlo = Whi + wplane;

    hipLaunchKernelGGL(split_x_kernel, dim3((unsigned)(xplane / 1024)), dim3(256),
                       0, stream, X, Xhi, Xlo);
    hipLaunchKernelGGL(build_w_kernel, dim3((unsigned)(wplane / 1024)), dim3(256),
                       0, stream, W, Whi, Wlo);
    hipLaunchKernelGGL(hc_gemm_kernel, dim3(N_TOTAL / BN, M_TOTAL / BM), dim3(256),
                       0, stream, Xhi, Xlo, Whi, Wlo, bias, Out);
}
